// OrdinalIPAttnProcessor2_0_9096740733327
// MI455X (gfx1250) — compile-verified, matrix operand order checked by reading, not run on hardware
//
#include <hip/hip_runtime.h>
#include <hip/hip_bf16.h>
#include <math.h>

// ---------------------------------------------------------------------------
// Types for CDNA5 WMMA (wave32): v_wmma_f32_16x16x32_bf16
// ---------------------------------------------------------------------------
typedef __attribute__((ext_vector_type(16))) __bf16    v16bf;
typedef __attribute__((ext_vector_type(8)))  __bf16    v8bf;
typedef __attribute__((ext_vector_type(8)))  float     v8f;
typedef __attribute__((ext_vector_type(4)))  unsigned  v4u;
typedef __attribute__((ext_vector_type(8)))  unsigned  v8u;

#define WMMA_BF16(A_, B_, C_) \
  __builtin_amdgcn_wmma_f32_16x16x32_bf16(false, (A_), false, (B_), (short)0, (C_), false, false)

static __device__ inline __bf16 f2bf(float f) {
  unsigned u = __float_as_uint(f);
  unsigned r = (u + 0x7FFFu + ((u >> 16) & 1u)) >> 16;   // round-to-nearest-even
  unsigned short s = (unsigned short)r;
  return __builtin_bit_cast(__bf16, s);
}

static __device__ inline v8bf zero_v8bf() {
  union { uint4 u; v8bf v; } z;
  z.u = make_uint4(0u, 0u, 0u, 0u);
  return z.v;
}

static __device__ inline v16bf cat8(v8bf lo, v8bf hi) {
  v16bf r;
#pragma unroll
  for (int i = 0; i < 8; ++i) { r[i] = lo[i]; r[i + 8] = hi[i]; }
  return r;
}

// ---------------------------------------------------------------------------
// Tensor Data Mover: 2D tile (bf16 elements) global -> LDS.
// D# per CDNA5 ISA ch.8: group0 = count/lds_addr/global_addr/type,
// group1 = data_size, pad, tensor dims (remaining extent => OOB rows read 0),
// tile dims, dim0 stride. pad16_4: +4 DWORDs after every 16 DWORDs, matching
// a 40-element bf16 LDS row pitch for a 32-element tile row.
// ---------------------------------------------------------------------------
static __device__ inline void tdm_load_2d_bf16(unsigned lds_byte, const void* gptr,
                                               unsigned dim0, unsigned dim1,
                                               unsigned tile0, unsigned tile1,
                                               unsigned stride0, bool pad16_4) {
  unsigned long long ga = (unsigned long long)(size_t)gptr;
  v4u g0;
  g0[0] = 1u;                                     // count=1, user descriptor
  g0[1] = lds_byte;                               // LDS byte address
  g0[2] = (unsigned)ga;                           // global_addr[31:0]
  g0[3] = (unsigned)(ga >> 32) | 0x80000000u;     // global_addr[56:32] | type=2
  v8u g1;
  g1[0] = 0x10000u |                              // data_size=1 (2 bytes), wg_mask=0
          (pad16_4 ? ((1u << 20) | (3u << 22) | (3u << 25)) : 0u);
  g1[1] = (dim0 & 0xFFFFu) << 16;                 // tensor_dim0[15:0]
  g1[2] = (dim0 >> 16) | ((dim1 & 0xFFFFu) << 16);// tensor_dim0[31:16] | tensor_dim1[15:0]
  g1[3] = (dim1 >> 16) | (tile0 << 16);           // tensor_dim1[31:16] | tile_dim0
  g1[4] = tile1;                                  // tile_dim1 (tile_dim2=0)
  g1[5] = stride0;                                // tensor_dim0_stride[31:0]
  g1[6] = 0u;
  g1[7] = 0u;
  asm volatile("tensor_load_to_lds %0, %1" :: "s"(g0), "s"(g1) : "memory");
}

static __device__ inline unsigned lds_addr_of(const void* p) {
  return (unsigned)(size_t)p;  // generic LDS ptr: low 32 bits = LDS byte offset
}

// ---------------------------------------------------------------------------
// One-time conversions: fp32 -> bf16 copy, and fp32 -> bf16 transpose (weights)
// ---------------------------------------------------------------------------
__global__ __launch_bounds__(256)
void cvt_bf16_kernel(const float* __restrict__ in, __bf16* __restrict__ out, long n) {
  long i = ((long)blockIdx.x * 256 + threadIdx.x) * 4;
  if (i < n) {
    float4 v = *(const float4*)(in + i);
    out[i + 0] = f2bf(v.x); out[i + 1] = f2bf(v.y);
    out[i + 2] = f2bf(v.z); out[i + 3] = f2bf(v.w);
  }
}

__global__ __launch_bounds__(256)
void transpose_bf16_kernel(const float* __restrict__ W, __bf16* __restrict__ Wt,
                           int K, int N) {  // Wt[n][k] = W[k][n]
  __shared__ float t[32][33];
  int k0 = blockIdx.x * 32, n0 = blockIdx.y * 32;
  int tx = threadIdx.x & 31, ty = threadIdx.x >> 5;     // 8 row-groups
  for (int i = ty; i < 32; i += 8) t[i][tx] = W[(size_t)(k0 + i) * N + n0 + tx];
  __syncthreads();
  for (int i = ty; i < 32; i += 8)
    Wt[(size_t)(n0 + i) * K + k0 + tx] = f2bf(t[tx][i]);
}

// ---------------------------------------------------------------------------
// GEMM: Out[M,N] = A[M,K](bf16) * Wt[N,K](bf16, pre-transposed)
//       (+ bias + residual, fp32 out, when FINAL)
// 256 thr = 8 waves; tile 128x128, K-step 32. Both tiles staged by TDM,
// double-buffered: tile i+1's DMA overlaps tile i's WMMAs
// (TDM ops of one wave complete in order; s_wait_tensorcnt 2 leaves the
// two prefetch descriptors in flight).
// ---------------------------------------------------------------------------
template <bool FINAL>
__global__ __launch_bounds__(256)
void gemm_tdm_kernel(const __bf16* __restrict__ A, const __bf16* __restrict__ Wt,
                     const float* __restrict__ bias, const float* __restrict__ resid,
                     void* __restrict__ Out, int M, int K, int N) {
  __shared__ __bf16 As[2][128][40];   // 32 used/row; 40-pitch == TDM pad 16dw->+4dw
  __shared__ __bf16 Bs[2][128][40];   // Bs[buf][n][k]

  const int tid = threadIdx.x;
  const int lane = tid & 31, wid = tid >> 5;
  const int lrow = lane & 15, half = lane >> 4;
  const int klane = half * 8;
  const int m0 = blockIdx.x * 128;
  const int n0 = blockIdx.y * 128;
  const int wm = (wid >> 1) * 32;
  const int wn = (wid & 1) * 64;

  v8f acc[2][4];
#pragma unroll
  for (int mi = 0; mi < 2; ++mi)
#pragma unroll
    for (int ni = 0; ni < 4; ++ni) acc[mi][ni] = (v8f){};

  const int ntiles = K >> 5;   // K % 32 == 0 for all uses

  if (wid == 0) {
    // prologue: tile 0 into buffer 0
    tdm_load_2d_bf16(lds_addr_of(&As[0][0][0]), A + (size_t)m0 * K,
                     (unsigned)K, (unsigned)(M - m0), 32u, 128u, (unsigned)K, true);
    tdm_load_2d_bf16(lds_addr_of(&Bs[0][0][0]), Wt + (size_t)n0 * K,
                     (unsigned)K, (unsigned)(N - n0), 32u, 128u, (unsigned)K, true);
  }

  for (int i = 0; i < ntiles; ++i) {
    const int cur = i & 1, nxt = cur ^ 1;
    if (wid == 0) {
      if (i + 1 < ntiles) {
        int k0 = (i + 1) << 5;
        tdm_load_2d_bf16(lds_addr_of(&As[nxt][0][0]), A + (size_t)m0 * K + k0,
                         (unsigned)(K - k0), (unsigned)(M - m0),
                         32u, 128u, (unsigned)K, true);
        tdm_load_2d_bf16(lds_addr_of(&Bs[nxt][0][0]), Wt + (size_t)n0 * K + k0,
                         (unsigned)(K - k0), (unsigned)(N - n0),
                         32u, 128u, (unsigned)K, true);
        __builtin_amdgcn_s_wait_tensorcnt(2);   // tile i done; tile i+1 in flight
      } else {
        __builtin_amdgcn_s_wait_tensorcnt(0);   // last tile done
      }
    }
    __syncthreads();

    v16bf afr[2], bfr[4];
#pragma unroll
    for (int mi = 0; mi < 2; ++mi) {
      int r = wm + mi * 16 + lrow;
      afr[mi] = cat8(*(const v8bf*)&As[cur][r][klane],
                     *(const v8bf*)&As[cur][r][klane + 16]);
    }
#pragma unroll
    for (int ni = 0; ni < 4; ++ni) {
      int c = wn + ni * 16 + lrow;
      bfr[ni] = cat8(*(const v8bf*)&Bs[cur][c][klane],
                     *(const v8bf*)&Bs[cur][c][klane + 16]);
    }
#pragma unroll
    for (int mi = 0; mi < 2; ++mi)
#pragma unroll
      for (int ni = 0; ni < 4; ++ni)
        acc[mi][ni] = WMMA_BF16(afr[mi], bfr[ni], acc[mi][ni]);
    __syncthreads();   // reads of buf 'cur' done before tile i+2 overwrites it
  }

  // D layout: lanes 0-15 -> M=v, lanes 16-31 -> M=v+8; N = lrow
#pragma unroll
  for (int mi = 0; mi < 2; ++mi)
#pragma unroll
    for (int ni = 0; ni < 4; ++ni)
#pragma unroll
      for (int v = 0; v < 8; ++v) {
        int row = m0 + wm + mi * 16 + v + 8 * half;
        int col = n0 + wn + ni * 16 + lrow;
        if (row < M) {
          if (FINAL) {
            float* O = (float*)Out;
            O[(size_t)row * N + col] =
                acc[mi][ni][v] + bias[col] + resid[(size_t)row * N + col];
          } else {
            __bf16* O = (__bf16*)Out;
            O[(size_t)row * N + col] = f2bf(acc[mi][ni][v]);
          }
        }
      }
}

// ---------------------------------------------------------------------------
// Attention: per (b, head): scores = Q K^T / 8, softmax over 109 keys, x V.
// 128 thr = 4 waves; each wave owns 16 q-rows; keys padded to 128 in LDS
// (TDM zero-fills rows >= Skv via tensor_dim1 = Skv). The K-tile TDM
// overlaps the manual V-transpose staging.
// ---------------------------------------------------------------------------
#define N_HEADS 20
#define SKV     109
#define HDIM    64
#define SKVP    128

__global__ __launch_bounds__(128)
void attn_kernel(const __bf16* __restrict__ Q, const __bf16* __restrict__ Kb,
                 const __bf16* __restrict__ Vb, __bf16* __restrict__ O,
                 int Sq, int H) {
  __shared__ __bf16 Ks[SKVP][HDIM];      // [key][d]       16 KB (TDM-staged)
  __shared__ __bf16 Vs[HDIM][SKVP];      // [d][key]       16 KB (transposed)
  __shared__ __bf16 Ps[4][16][SKVP];     // per-wave probs 16 KB

  const int tid = threadIdx.x, lane = tid & 31, wid = tid >> 5;
  const int lrow = lane & 15, half = lane >> 4, klane = half * 8;
  const int bh = blockIdx.y, b = bh / N_HEADS, h = bh % N_HEADS;
  const size_t kvbase = (size_t)b * SKV * H + (size_t)h * HDIM;

  // K tile via TDM: 128x64 @ row 0, col h*64; rows >= SKV zero-filled
  if (wid == 0) {
    tdm_load_2d_bf16(lds_addr_of(&Ks[0][0]), Kb + kvbase,
                     (unsigned)(H - h * HDIM), (unsigned)SKV,
                     (unsigned)HDIM, (unsigned)SKVP, (unsigned)H, false);
  }
  // V staged manually (needs transpose into Vs[d][key]) -- overlaps K DMA
  for (int i = tid; i < SKVP * (HDIM / 8); i += 128) {
    int n = i >> 3, c = (i & 7) * 8;
    v8bf v = zero_v8bf();
    if (n < SKV) v = *(const v8bf*)(Vb + kvbase + (size_t)n * H + c);
#pragma unroll
    for (int j = 0; j < 8; ++j) Vs[c + j][n] = v[j];
  }
  if (wid == 0) __builtin_amdgcn_s_wait_tensorcnt(0);
  __syncthreads();

  // Q fragments for this wave's 16 rows (A-frag layout over d)
  const int srow0 = blockIdx.x * 64 + wid * 16;
  const size_t qbase = ((size_t)(b * Sq + srow0 + lrow)) * H + (size_t)h * HDIM;
  v16bf qa[2];
#pragma unroll
  for (int kb = 0; kb < 2; ++kb)
    qa[kb] = cat8(*(const v8bf*)(Q + qbase + kb * 32 + klane),
                  *(const v8bf*)(Q + qbase + kb * 32 + klane + 16));

  // scores: 8 key-tiles of 16, d=64 -> 2 WMMAs each
  v8f sc[8];
#pragma unroll
  for (int t = 0; t < 8; ++t) {
    int n = t * 16 + lrow;
    v16bf k0 = cat8(*(const v8bf*)&Ks[n][klane],      *(const v8bf*)&Ks[n][klane + 16]);
    v16bf k1 = cat8(*(const v8bf*)&Ks[n][32 + klane], *(const v8bf*)&Ks[n][32 + klane + 16]);
    v8f c = (v8f){};
    c = WMMA_BF16(qa[0], k0, c);
    c = WMMA_BF16(qa[1], k1, c);
    sc[t] = c;
  }

  // scale + key-mask + softmax (rows in VGPR index, key cols in lanes 0-15)
  const float NEG = -__builtin_inff();
  float rmax[8], rsum[8];
#pragma unroll
  for (int v = 0; v < 8; ++v) rmax[v] = NEG;
#pragma unroll
  for (int t = 0; t < 8; ++t) {
    bool valid = (t * 16 + lrow) < SKV;
#pragma unroll
    for (int v = 0; v < 8; ++v) {
      float s = valid ? sc[t][v] * 0.125f : NEG;   // 1/sqrt(64)
      sc[t][v] = s;
      rmax[v] = fmaxf(rmax[v], s);
    }
  }
#pragma unroll
  for (int off = 1; off < 16; off <<= 1)
#pragma unroll
    for (int v = 0; v < 8; ++v)
      rmax[v] = fmaxf(rmax[v], __shfl_xor(rmax[v], off, 32));
#pragma unroll
  for (int v = 0; v < 8; ++v) rsum[v] = 0.f;
#pragma unroll
  for (int t = 0; t < 8; ++t)
#pragma unroll
    for (int v = 0; v < 8; ++v) {
      float p = __expf(sc[t][v] - rmax[v]);
      sc[t][v] = p;
      rsum[v] += p;
    }
#pragma unroll
  for (int off = 1; off < 16; off <<= 1)
#pragma unroll
    for (int v = 0; v < 8; ++v)
      rsum[v] += __shfl_xor(rsum[v], off, 32);

  // probs -> bf16 -> LDS, re-read in A-fragment layout
#pragma unroll
  for (int t = 0; t < 8; ++t)
#pragma unroll
    for (int v = 0; v < 8; ++v)
      Ps[wid][v + 8 * half][t * 16 + lrow] = f2bf(sc[t][v] * (1.f / rsum[v]));
  __syncthreads();

  // out = P x V : 4 d-tiles x 4 key-steps of 32 = 16 WMMAs
  v8f o[4];
#pragma unroll
  for (int dt = 0; dt < 4; ++dt) o[dt] = (v8f){};
#pragma unroll
  for (int kb = 0; kb < SKVP; kb += 32) {
    v16bf pf = cat8(*(const v8bf*)&Ps[wid][lrow][kb + klane],
                    *(const v8bf*)&Ps[wid][lrow][kb + klane + 16]);
#pragma unroll
    for (int dt = 0; dt < 4; ++dt) {
      int d = dt * 16 + lrow;
      v16bf vf = cat8(*(const v8bf*)&Vs[d][kb + klane],
                      *(const v8bf*)&Vs[d][kb + klane + 16]);
      o[dt] = WMMA_BF16(pf, vf, o[dt]);
    }
  }

#pragma unroll
  for (int dt = 0; dt < 4; ++dt)
#pragma unroll
    for (int v = 0; v < 8; ++v) {
      int srow = srow0 + v + 8 * half;
      O[((size_t)(b * Sq + srow)) * H + (size_t)h * HDIM + dt * 16 + lrow] =
          f2bf(o[dt][v]);
    }
}

// ---------------------------------------------------------------------------
// Launcher
// ---------------------------------------------------------------------------
extern "C" void kernel_launch(void* const* d_in, const int* in_sizes, int n_in,
                              void* d_out, int out_size, void* d_ws, size_t ws_size,
                              hipStream_t stream) {
  (void)in_sizes; (void)n_in; (void)out_size; (void)ws_size;
  const float* HS  = (const float*)d_in[0];  // (8, 4096, 1280)
  const float* EHS = (const float*)d_in[1];  // (8, 109, 768)
  const float* Wq  = (const float*)d_in[2];  // (1280, 1280)
  const float* Wk  = (const float*)d_in[3];  // (768, 1280)
  const float* Wv  = (const float*)d_in[4];  // (768, 1280)
  const float* Wo  = (const float*)d_in[5];  // (1280, 1280)
  const float* bo  = (const float*)d_in[6];  // (1280,)
  float* out = (float*)d_out;

  const int B = 8, Sq = 4096, H = 1280, C = 768, Skv = 109;
  const int M   = B * Sq;    // 32768
  const int Mkv = B * Skv;   // 872

  // workspace layout (bf16 elements)
  __bf16* HSb  = (__bf16*)d_ws;
  __bf16* EHSb = HSb  + (size_t)M * H;
  __bf16* WqT  = EHSb + (size_t)Mkv * C;
  __bf16* WkT  = WqT  + (size_t)H * H;
  __bf16* WvT  = WkT  + (size_t)H * C;     // Wt[N=H][K=C]
  __bf16* WoT  = WvT  + (size_t)H * C;
  __bf16* Qb   = WoT  + (size_t)H * H;
  __bf16* Kb   = Qb   + (size_t)M * H;
  __bf16* Vb   = Kb   + (size_t)Mkv * H;
  __bf16* Ob   = Vb   + (size_t)Mkv * H;

  dim3 blk(256);
  // one-time conversions / weight transposes
  {
    long nHS = (long)M * H, nE = (long)Mkv * C;
    cvt_bf16_kernel<<<dim3((unsigned)((nHS / 4 + 255) / 256)), blk, 0, stream>>>(HS, HSb, nHS);
    cvt_bf16_kernel<<<dim3((unsigned)((nE  / 4 + 255) / 256)), blk, 0, stream>>>(EHS, EHSb, nE);
    transpose_bf16_kernel<<<dim3(H / 32, H / 32), blk, 0, stream>>>(Wq, WqT, H, H);
    transpose_bf16_kernel<<<dim3(C / 32, H / 32), blk, 0, stream>>>(Wk, WkT, C, H);
    transpose_bf16_kernel<<<dim3(C / 32, H / 32), blk, 0, stream>>>(Wv, WvT, C, H);
    transpose_bf16_kernel<<<dim3(H / 32, H / 32), blk, 0, stream>>>(Wo, WoT, H, H);
  }
  // Q = HS @ Wq, K = EHS @ Wk, V = EHS @ Wv  (bf16 out)
  gemm_tdm_kernel<false><<<dim3(M / 128, H / 128), blk, 0, stream>>>(
      HSb, WqT, nullptr, nullptr, Qb, M, H, H);
  gemm_tdm_kernel<false><<<dim3((Mkv + 127) / 128, H / 128), blk, 0, stream>>>(
      EHSb, WkT, nullptr, nullptr, Kb, Mkv, C, H);
  gemm_tdm_kernel<false><<<dim3((Mkv + 127) / 128, H / 128), blk, 0, stream>>>(
      EHSb, WvT, nullptr, nullptr, Vb, Mkv, C, H);
  // attention per (b, h), 64 q-rows per block
  attn_kernel<<<dim3(Sq / 64, B * N_HEADS), dim3(128), 0, stream>>>(
      Qb, Kb, Vb, Ob, Sq, H);
  // out = O @ Wo + bo + HS
  gemm_tdm_kernel<true><<<dim3(M / 128, H / 128), blk, 0, stream>>>(
      Ob, WoT, bo, HS, out, M, H, H);
}
